// TextRNN_91001767068107
// MI455X (gfx1250) — matrix operand order repaired and checked by reading, not compiled, not run on hardware
//
#include <hip/hip_runtime.h>
#include <hip/hip_bf16.h>

typedef __attribute__((ext_vector_type(16))) __bf16 v16bf;
typedef __attribute__((ext_vector_type(8)))  float  v8f;
typedef __attribute__((ext_vector_type(4)))  unsigned int u32x4;
typedef __attribute__((ext_vector_type(4)))  int   i32x4;
typedef __attribute__((ext_vector_type(8)))  int   i32x8;

#define T_LEN 512
#define B_SZ  128
#define E_DIM 256
#define H_DIM 256
#define G4H   1024
#define C_CLS 20

// xw fragment-native layout: [t][mt(8)][nt(16)][gate(4)][lane(32)][r(8)] bf16
// slice(t,mt) = 16*4*32*8 = 16384 bf16 = 32 KB, contiguous.
#define SLICE_ELEMS 16384

__device__ __forceinline__ float sigmoidf_(float x) {
    return 1.0f / (1.0f + __expf(-x));
}

// ---------------------------------------------------------------------------
// Kernel 1: W_f, U_f, W_b, U_b ([256,1024] f32) -> bf16 transposed [1024,256]
// so WMMA B-fragments (lane = N, j -> K = 16*hi + j) are contiguous per lane.
// ---------------------------------------------------------------------------
__global__ void cvt_transpose_kernel(const float* __restrict__ Wf,
                                     const float* __restrict__ Uf,
                                     const float* __restrict__ Wb,
                                     const float* __restrict__ Ub,
                                     __bf16* __restrict__ out) {
    int idx = blockIdx.x * blockDim.x + threadIdx.x;   // 4 * 262144 total
    int which = idx >> 18;
    int rem   = idx & 262143;
    int n = rem >> 8;
    int k = rem & 255;
    const float* src = (which == 0) ? Wf : (which == 1) ? Uf
                     : (which == 2) ? Wb : Ub;
    out[which * 262144 + n * 256 + k] = (__bf16)src[k * 1024 + n];
}

// ---------------------------------------------------------------------------
// Kernel 2: fused embedding-gather + input projection + bias, stored in
// fragment-native layout (one b128 store per lane per tile).
// grid = (16, 2048, 2), block = 256 (8 waves, one 16x16 tile each).
// ---------------------------------------------------------------------------
__global__ __launch_bounds__(256) void proj_kernel(
    const int*   __restrict__ tokens,
    const float* __restrict__ emb,
    const __bf16* __restrict__ Wt_f, const __bf16* __restrict__ Wt_b,
    const float* __restrict__ b_f,   const float* __restrict__ b_b,
    __bf16* __restrict__ xw_f,       __bf16* __restrict__ xw_b) {

    const int dir  = blockIdx.z;
    const int lane = threadIdx.x & 31;
    const int w    = threadIdx.x >> 5;
    const int mtg  = blockIdx.y * 2 + (w >> 2);   // global M tile 0..4095
    const int ntg  = blockIdx.x * 4 + (w & 3);    // global N tile 0..63
    const int hi   = lane >> 4;
    const int kb   = 8 * hi;

    // A layout: lane = M row of the flattened [T,B] activation matrix
    const int mrow = mtg * 16 + (lane & 15);
    const int t    = mrow >> 7;
    const int b    = mrow & 127;
    const int tcol = dir ? (T_LEN - 1 - t) : t;
    const int tok  = tokens[b * T_LEN + tcol];
    const float* arow = emb + (long)tok * E_DIM;

    const __bf16* Wt   = dir ? Wt_b : Wt_f;
    const float*  bias = dir ? b_b  : b_f;
    __bf16*       xw   = dir ? xw_b : xw_f;

    const int ncol = ntg * 16 + (lane & 15);      // B layout: lane = N

    v8f acc = {};
    for (int kk = 0; kk < 256; kk += 32) {
        float4 a0 = *(const float4*)(arow + kk + 16 * hi);
        float4 a1 = *(const float4*)(arow + kk + 16 * hi + 4);
        float4 a2 = *(const float4*)(arow + kk + 16 * hi + 8);
        float4 a3 = *(const float4*)(arow + kk + 16 * hi + 12);
        v16bf a;
        a[0]  = (__bf16)a0.x; a[1]  = (__bf16)a0.y; a[2]  = (__bf16)a0.z; a[3]  = (__bf16)a0.w;
        a[4]  = (__bf16)a1.x; a[5]  = (__bf16)a1.y; a[6]  = (__bf16)a1.z; a[7]  = (__bf16)a1.w;
        a[8]  = (__bf16)a2.x; a[9]  = (__bf16)a2.y; a[10] = (__bf16)a2.z; a[11] = (__bf16)a2.w;
        a[12] = (__bf16)a3.x; a[13] = (__bf16)a3.y; a[14] = (__bf16)a3.z; a[15] = (__bf16)a3.w;

        v16bf bm = *(const v16bf*)(Wt + (long)ncol * 256 + kk + 16 * hi);

        acc = __builtin_amdgcn_wmma_f32_16x16x32_bf16(
                  false, a, false, bm, (short)0, acc, false, false);
    }

    // Fragment-native store: tile coords (t, mtb, nt, gate); lane keeps its
    // 8 D values contiguous -> single 16B store.
    const int   mtb  = mtg & 7;          // batch slice (t = mtg >> 3)
    const int   gate = ntg >> 4;
    const int   nt   = ntg & 15;
    const float bv   = bias[ncol];
    union { __bf16 h[8]; uint4 u; } st;
#pragma unroll
    for (int r = 0; r < 8; ++r) st.h[r] = (__bf16)(acc[r] + bv);
    long off = ((((long)(mtg >> 3) * 8 + mtb) * 16 + nt) * 4 + gate) * 256 + lane * 8;
    *(uint4*)(xw + off) = st.u;
}

// ---------------------------------------------------------------------------
// Kernel 3: LSTM scan. 2 dirs x 8 batch slices of 16 rows; h-slice [16,256]
// bf16 recirculates in LDS, c in VGPRs. Per-step 32KB xw slice is staged
// LDS-double-buffered by the Tensor Data Mover (TDM), overlapped with WMMA.
// grid = (2, 8), block = 512 (16 waves = one H-col tile each).
// Dynamic LDS: [h: 8KB][xw buf0: 32KB][xw buf1: 32KB] = 72KB.
// ---------------------------------------------------------------------------
__device__ __forceinline__ void tdm_load_slice(const __bf16* gsrc,
                                               unsigned lds_byte_addr) {
    unsigned long long ga = (unsigned long long)(uintptr_t)gsrc;
    // D# group 0: count=1, lds_addr, global_addr, type=2
    u32x4 g0;
    g0[0] = 1u;                                  // count=1, user mode
    g0[1] = lds_byte_addr;                       // LDS dest (bytes)
    g0[2] = (unsigned)(ga & 0xFFFFFFFFull);      // global_addr[31:0]
    g0[3] = (unsigned)((ga >> 32) & 0x01FFFFFFull) | (2u << 30);  // [56:32] | type=2
    // D# group 1: data_size=8B, tensor_dim0=4096, tensor_dim1=1,
    // tile_dim0=4096, tile_dim1=1, dim0_stride=4096
    i32x8 g1;
    g1[0] = (int)(3u << 16);        // workgroup_mask=0, data_size=3 (8B)
    g1[1] = (int)(4096u << 16);     // tensor_dim0[15:0] in [31:16]
    g1[2] = (int)(1u << 16);        // tensor_dim0 hi=0, tensor_dim1 lo=1
    g1[3] = (int)(4096u << 16);     // tensor_dim1 hi=0, tile_dim0=4096
    g1[4] = 1;                      // tile_dim1=1, tile_dim2=0
    g1[5] = 4096;                   // tensor_dim0_stride lo
    g1[6] = 0;
    g1[7] = 0;
    i32x4 zz = {0, 0, 0, 0};
#if defined(__clang_major__) && (__clang_major__ >= 23)
    i32x8 z8 = {0, 0, 0, 0, 0, 0, 0, 0};
    __builtin_amdgcn_tensor_load_to_lds(g0, g1, zz, zz, z8, 0);
#else
    __builtin_amdgcn_tensor_load_to_lds(g0, g1, zz, zz, 0);
#endif
}

__global__ __launch_bounds__(512) void lstm_scan_kernel(
    const __bf16* __restrict__ xw_f, const __bf16* __restrict__ xw_b,
    const __bf16* __restrict__ Ut_f, const __bf16* __restrict__ Ut_b,
    float* __restrict__ hout_f,      float* __restrict__ hout_b) {

    extern __shared__ __bf16 smem[];
    __bf16* h_lds = smem;                 // 16*256 = 4096 elems (8KB)
    __bf16* zbuf0 = smem + 4096;          // SLICE_ELEMS (32KB)
    __bf16* zbuf1 = zbuf0 + SLICE_ELEMS;  // SLICE_ELEMS (32KB)

    const int dir = blockIdx.x;
    const int mt  = blockIdx.y;           // batch slice 0..7
    const __bf16* xw = dir ? xw_b : xw_f;
    const __bf16* Ut = dir ? Ut_b : Ut_f;
    float* hout = dir ? hout_b : hout_f;

    const int tid  = threadIdx.x;
    const int lane = tid & 31;
    const int w    = tid >> 5;            // wave = H-col tile 0..15
    const int hi   = lane >> 4;
    const int nt   = w;
    const int mloc = lane & 15;
    const int nch  = nt * 16 + (lane & 15);

    for (int i = tid; i < 16 * 256; i += 512) h_lds[i] = (__bf16)0.0f;

    float c[8], hnew[8];
#pragma unroll
    for (int r = 0; r < 8; ++r) c[r] = 0.0f;

    const unsigned zb0 = (unsigned)(uintptr_t)zbuf0;
    const unsigned zb1 = (unsigned)(uintptr_t)zbuf1;

    // Prologue: stage t=0 slice via TDM
    if (w == 0) {
        tdm_load_slice(xw + (long)mt * SLICE_ELEMS, zb0);
        __builtin_amdgcn_s_wait_tensorcnt(0);
    }
    __syncthreads();

    int cur = 0;
    for (int t = 0; t < T_LEN; ++t) {
        // Kick off TDM for step t+1 into the other buffer (overlaps WMMAs)
        if (w == 0 && t + 1 < T_LEN) {
            tdm_load_slice(xw + ((long)(t + 1) * 8 + mt) * SLICE_ELEMS,
                           cur ? zb0 : zb1);
        }

        v8f zi = {}, zf = {}, zg = {}, zo = {};
        for (int kk = 0; kk < 256; kk += 32) {
            union { uint4 u[2]; v16bf v; } cv;
            cv.u[0] = *(const uint4*)&h_lds[mloc * 256 + kk + 16 * hi];
            cv.u[1] = *(const uint4*)&h_lds[mloc * 256 + kk + 16 * hi + 8];
            v16bf a = cv.v;

            const __bf16* ub = Ut + kk + 16 * hi;
            v16bf b0 = *(const v16bf*)(ub + (long)(0   + nch) * 256);
            v16bf b1 = *(const v16bf*)(ub + (long)(256 + nch) * 256);
            v16bf b2 = *(const v16bf*)(ub + (long)(512 + nch) * 256);
            v16bf b3 = *(const v16bf*)(ub + (long)(768 + nch) * 256);

            zi = __builtin_amdgcn_wmma_f32_16x16x32_bf16(false, a, false, b0, (short)0, zi, false, false);
            zf = __builtin_amdgcn_wmma_f32_16x16x32_bf16(false, a, false, b1, (short)0, zf, false, false);
            zg = __builtin_amdgcn_wmma_f32_16x16x32_bf16(false, a, false, b2, (short)0, zg, false, false);
            zo = __builtin_amdgcn_wmma_f32_16x16x32_bf16(false, a, false, b3, (short)0, zo, false, false);
        }

        // Gate adds from LDS-staged slice: fragment-identity, one 16B load/gate
        const __bf16* zb = cur ? zbuf1 : zbuf0;
        union { uint4 u; __bf16 h[8]; } gi, gf, gg, go;
        gi.u = *(const uint4*)(zb + ((nt * 4 + 0) * 32 + lane) * 8);
        gf.u = *(const uint4*)(zb + ((nt * 4 + 1) * 32 + lane) * 8);
        gg.u = *(const uint4*)(zb + ((nt * 4 + 2) * 32 + lane) * 8);
        go.u = *(const uint4*)(zb + ((nt * 4 + 3) * 32 + lane) * 8);

#pragma unroll
        for (int r = 0; r < 8; ++r) {
            float vi = zi[r] + (float)gi.h[r];
            float vf = zf[r] + (float)gf.h[r];
            float vg = zg[r] + (float)gg.h[r];
            float vo = zo[r] + (float)go.h[r];
            float ig = sigmoidf_(vi);
            float fg = sigmoidf_(vf);
            float ga = tanhf(vg);
            float og = sigmoidf_(vo);
            float cc = fg * c[r] + ig * ga;
            c[r]     = cc;
            hnew[r]  = og * tanhf(cc);
        }

        if (w == 0) __builtin_amdgcn_s_wait_tensorcnt(0);  // next slice landed
        __syncthreads();   // all h reads + TDM completion visible
#pragma unroll
        for (int r = 0; r < 8; ++r)
            h_lds[(r + 8 * hi) * 256 + nch] = (__bf16)hnew[r];
        __syncthreads();   // new h visible to all waves
        cur ^= 1;
    }

    for (int i = tid; i < 16 * 256; i += 512)
        hout[mt * 4096 + i] = (float)h_lds[i];
}

// ---------------------------------------------------------------------------
// Kernel 4: dense [128,512]@[512,20] + bias + softmax. Tiny; one thread/row.
// ---------------------------------------------------------------------------
__global__ void dense_softmax_kernel(const float* __restrict__ hf,
                                     const float* __restrict__ hb,
                                     const float* __restrict__ Wd,
                                     const float* __restrict__ bd,
                                     float* __restrict__ out) {
    int b = threadIdx.x;   // 0..127
    float acc[C_CLS];
#pragma unroll
    for (int cix = 0; cix < C_CLS; ++cix) acc[cix] = bd[cix];
    for (int k = 0; k < H_DIM; ++k) {
        float y = hf[b * H_DIM + k];
        const float* wr = Wd + k * C_CLS;
#pragma unroll
        for (int cix = 0; cix < C_CLS; ++cix) acc[cix] += y * wr[cix];
    }
    for (int k = 0; k < H_DIM; ++k) {
        float y = hb[b * H_DIM + k];
        const float* wr = Wd + (H_DIM + k) * C_CLS;
#pragma unroll
        for (int cix = 0; cix < C_CLS; ++cix) acc[cix] += y * wr[cix];
    }
    float mx = acc[0];
#pragma unroll
    for (int cix = 1; cix < C_CLS; ++cix) mx = fmaxf(mx, acc[cix]);
    float sum = 0.0f;
#pragma unroll
    for (int cix = 0; cix < C_CLS; ++cix) {
        acc[cix] = __expf(acc[cix] - mx);
        sum += acc[cix];
    }
    float inv = 1.0f / sum;
#pragma unroll
    for (int cix = 0; cix < C_CLS; ++cix)
        out[b * C_CLS + cix] = acc[cix] * inv;
}

// ---------------------------------------------------------------------------
extern "C" void kernel_launch(void* const* d_in, const int* in_sizes, int n_in,
                              void* d_out, int out_size, void* d_ws, size_t ws_size,
                              hipStream_t stream) {
    const int*   tokens = (const int*)  d_in[0];
    const float* emb    = (const float*)d_in[1];
    const float* W_f    = (const float*)d_in[2];
    const float* U_f    = (const float*)d_in[3];
    const float* b_f    = (const float*)d_in[4];
    const float* W_b    = (const float*)d_in[5];
    const float* U_b    = (const float*)d_in[6];
    const float* b_b    = (const float*)d_in[7];
    const float* W_d    = (const float*)d_in[8];
    const float* b_d    = (const float*)d_in[9];

    __bf16* wsb  = (__bf16*)d_ws;
    __bf16* Wt_f = wsb;                       // 262144
    __bf16* Ut_f = wsb + 262144;
    __bf16* Wt_b = wsb + 524288;
    __bf16* Ut_b = wsb + 786432;
    __bf16* xw_f = wsb + 1048576;             // 67108864 (fragment-native)
    __bf16* xw_b = xw_f + 67108864;
    float*  hout_f = (float*)(xw_b + 67108864);
    float*  hout_b = hout_f + B_SZ * H_DIM;

    cvt_transpose_kernel<<<4096, 256, 0, stream>>>(W_f, U_f, W_b, U_b, wsb);

    proj_kernel<<<dim3(16, 2048, 2), 256, 0, stream>>>(
        tokens, emb, Wt_f, Wt_b, b_f, b_b, xw_f, xw_b);

    // 72KB dynamic LDS: h (8KB) + xw double buffer (2x32KB)
    lstm_scan_kernel<<<dim3(2, 8), 512, 73728, stream>>>(
        xw_f, xw_b, Ut_f, Ut_b, hout_f, hout_b);

    dense_softmax_kernel<<<1, 128, 0, stream>>>(
        hout_f, hout_b, W_d, b_d, (float*)d_out);
}
